// Seq2seq_26955214750201
// MI455X (gfx1250) — matrix operand order repaired and checked by reading, not compiled, not run on hardware
//
#include <hip/hip_runtime.h>

typedef __bf16 bf16;
typedef __attribute__((ext_vector_type(16))) __bf16 v16bf;
typedef __attribute__((ext_vector_type(8)))  __bf16 v8bf;
typedef __attribute__((ext_vector_type(8)))  float  v8f;

#define B_     64
#define T_     128
#define EMB_   64
#define UNITS_ 512
#define GATES_ 2048   /* 4*UNITS */
#define VOCAB_ 2000
#define VTILE_PAD_ 2048  /* 125 real 16-wide N tiles, padded to 128 */

__device__ __forceinline__ float sigmoidf_(float x){ return 1.0f/(1.0f + __expf(-x)); }

// clamped exp-based tanh: one v_exp_f32 instead of the libm tanhf expansion
__device__ __forceinline__ float fast_tanhf(float x){
    float xc = fminf(15.0f, fmaxf(-15.0f, x));
    float e  = __expf(2.0f * xc);
    return (e - 1.0f) / (e + 1.0f);
}

// A-fragment for v_wmma_f32_16x16x32_bf16 (wave32):
// lanes 0-15  : M=lane,    K [k0,k0+16)
// lanes 16-31 : M=lane-16, K [k0+16,k0+32)
__device__ __forceinline__ v16bf load_a_frag(const bf16* row, int k0, int lane){
    int hb = (lane >> 4) << 4;                     // 0 or 16
    v8bf lo = *(const v8bf*)(row + k0 + hb);
    v8bf hi = *(const v8bf*)(row + k0 + hb + 8);
    return __builtin_shufflevector(lo, hi, 0,1,2,3,4,5,6,7,8,9,10,11,12,13,14,15);
}

// B-fragment: lane = N (mod 16), lanes<16 hold K=k0..k0+15, lanes>=16 hold K=k0+16..k0+31.
// Requires the B matrix stored [N][K] (pre-transposed) -> 32 contiguous bytes per lane.
__device__ __forceinline__ v16bf load_b_frag(const bf16* rowN, int k0, int lane){
    return *(const v16bf*)(rowN + k0 + ((lane >> 4) << 4));
}

__device__ __forceinline__ v8f wmma_bf16(v16bf a, v16bf b, v8f c){
    return __builtin_amdgcn_wmma_f32_16x16x32_bf16(false, a, false, b, (short)0, c, false, false);
}

// ---------------------------------------------------------------- prep kernels
__global__ void k_convert_bf16(const float* __restrict__ src, bf16* __restrict__ dst, int n){
    int i = blockIdx.x * blockDim.x + threadIdx.x;
    if (i < n) dst[i] = (bf16)src[i];
}

// src: [R][C] f32 row-major -> dst: [C][R] bf16 (WMMA B-fragment friendly layout)
__global__ void k_transpose_bf16(const float* __restrict__ src, bf16* __restrict__ dst, int R, int C){
    int i = blockIdx.x * blockDim.x + threadIdx.x;
    if (i < R * C){
        int r = i / C, c = i - r * C;
        dst[(size_t)c * R + r] = (bf16)src[i];
    }
}

// H [B][T][U] bf16 -> Hu [B][U][T] bf16 (B-fragment layout for attention GEMM2)
__global__ void k_transpose_H(const bf16* __restrict__ H, bf16* __restrict__ Hu){
    int i = blockIdx.x * blockDim.x + threadIdx.x;
    if (i < B_ * T_ * UNITS_){
        int b = i / (T_ * UNITS_);
        int rem = i - b * T_ * UNITS_;
        int t = rem / UNITS_, u = rem - t * UNITS_;
        Hu[((size_t)b * UNITS_ + u) * T_ + t] = H[i];
    }
}

// ------------------------------------------------- xz = emb[ids] @ Wx + b  (fp32 out)
// grid (512 m-tiles, 32 n-chunks), block 128 (4 waves, 1 n-tile each).
// Two independent accumulators (K=64 -> 2 k-steps) to avoid the WMMA RAW NOP chain.
__global__ void __launch_bounds__(128) k_xz(const int* __restrict__ ids,
                                            const bf16* __restrict__ emb,   // [VOCAB][EMB]
                                            const bf16* __restrict__ Wxt,   // [GATES][EMB]
                                            const float* __restrict__ bias, // [GATES]
                                            float* __restrict__ xz){        // [B*T][GATES]
    int lane = threadIdx.x & 31, w = threadIdx.x >> 5;
    int ln = lane & 15, lh = lane >> 4;
    int mt = blockIdx.x;
    int n0 = (blockIdx.y * 4 + w) * 16;
    int mrow = mt * 16 + ln;
    int id = ids[mrow];
    const bf16* arow = emb + (size_t)id * EMB_;
    const bf16* brow = Wxt + (size_t)(n0 + ln) * EMB_;
    v8f z = {0.f,0.f,0.f,0.f,0.f,0.f,0.f,0.f};
    v8f acc0 = wmma_bf16(load_a_frag(arow, 0,  lane), load_b_frag(brow, 0,  lane), z);
    v8f acc1 = wmma_bf16(load_a_frag(arow, 32, lane), load_b_frag(brow, 32, lane), z);
    int nc = n0 + ln;
    float bv = bias[nc];
    int mbase = mt * 16 + (lh << 3);
#pragma unroll
    for (int r = 0; r < 8; ++r)
        xz[(size_t)(mbase + r) * GATES_ + nc] = acc0[r] + acc1[r] + bv;
}

// --------------------------------------------------------------- LSTM scan
// Single workgroup, 1024 threads = 32 waves on one WGP. Wave w owns within-gate
// columns u in [16w, 16w+16) across ALL four gates -> i/f/g/o local to the wave.
// h (bf16) in 64 KB LDS; c (f32) in an L2-resident global buffer, per-thread owned.
// 16 independent accumulators between dependent WMMAs -> no hazard NOPs.
__global__ void __launch_bounds__(1024, 1) k_lstm_scan(
        const float* __restrict__ xz,    // [B][T][GATES] f32
        const bf16*  __restrict__ Wht,   // [GATES][UNITS] bf16 (transposed Wh)
        const float* __restrict__ h0,    // [B][UNITS] or null (zeros)
        const float* __restrict__ c0,    // [B][UNITS] or null (zeros)
        float* __restrict__ cio,         // [B][UNITS] c state; holds final c on exit
        float* __restrict__ hout,        // [B][UNITS] final h
        bf16*  __restrict__ Hout,        // [B][T][UNITS]
        bf16*  __restrict__ Sshift){     // [B][T][UNITS] or null: row t+1 = h_t, row 0 = h0
    __shared__ bf16 lds_h[B_ * UNITS_];  // 64 KB
    int tid = threadIdx.x;
    for (int i = tid; i < B_ * UNITS_; i += 1024){
        float hv = h0 ? h0[i] : 0.0f;
        bf16 hb = (bf16)hv;
        lds_h[i] = hb;
        if (Sshift){
            int b = i >> 9, u = i & 511;
            Sshift[((size_t)b * T_) * UNITS_ + u] = hb;   // row t=0 = s0
        }
    }
    __syncthreads();

    int lane = tid & 31, w = tid >> 5;
    int ln = lane & 15, lh = lane >> 4;
    int u0 = w * 16;
    int u  = u0 + ln;

    for (int t = 0; t < T_; ++t){
        v8f acc[4][4];
#pragma unroll
        for (int mt = 0; mt < 4; ++mt)
#pragma unroll
            for (int g = 0; g < 4; ++g)
                acc[mt][g] = (v8f){0.f,0.f,0.f,0.f,0.f,0.f,0.f,0.f};

        for (int k0 = 0; k0 < UNITS_; k0 += 32){
            v16bf a[4];
#pragma unroll
            for (int mt = 0; mt < 4; ++mt)
                a[mt] = load_a_frag(lds_h + (mt * 16 + ln) * UNITS_, k0, lane);
#pragma unroll
            for (int g = 0; g < 4; ++g){
                const bf16* brow = Wht + (size_t)(g * UNITS_ + u0 + ln) * UNITS_;
                if (k0 + 32 < UNITS_) __builtin_prefetch(brow + k0 + 32 + (lh << 4), 0, 1);
                v16bf b = load_b_frag(brow, k0, lane);
#pragma unroll
                for (int mt = 0; mt < 4; ++mt)
                    acc[mt][g] = wmma_bf16(a[mt], b, acc[mt][g]);
            }
        }
        __syncthreads();   // all LDS h reads complete before overwrite

#pragma unroll
        for (int mt = 0; mt < 4; ++mt){
#pragma unroll
            for (int r = 0; r < 8; ++r){
                int m = mt * 16 + (lh << 3) + r;           // batch row
                const float* xzr = xz + ((size_t)m * T_ + t) * GATES_;
                float zi = acc[mt][0][r] + xzr[0 * UNITS_ + u];
                float zf = acc[mt][1][r] + xzr[1 * UNITS_ + u];
                float zg = acc[mt][2][r] + xzr[2 * UNITS_ + u];
                float zo = acc[mt][3][r] + xzr[3 * UNITS_ + u];
                float ig = sigmoidf_(zi), fg = sigmoidf_(zf);
                float gg = fast_tanhf(zg), og = sigmoidf_(zo);
                float cold = (t == 0) ? (c0 ? c0[m * UNITS_ + u] : 0.0f)
                                      : cio[m * UNITS_ + u];
                float cn = fg * cold + ig * gg;
                cio[m * UNITS_ + u] = cn;
                float hn = og * fast_tanhf(cn);
                bf16 hb = (bf16)hn;
                lds_h[m * UNITS_ + u] = hb;
                Hout[((size_t)m * T_ + t) * UNITS_ + u] = hb;
                if (Sshift && (t + 1) < T_)
                    Sshift[((size_t)m * T_ + (t + 1)) * UNITS_ + u] = hb;
                if (t == T_ - 1) hout[m * UNITS_ + u] = hn;
            }
        }
        __syncthreads();   // new h visible for next step
    }
}

// --------------------------------------------------------- fused attention (per batch)
// scores = S_ @ H^T -> softmax -> A = P @ H.  Block = 256 threads (8 waves), 1 batch.
__global__ void __launch_bounds__(256) k_attention(
        const bf16* __restrict__ Ssh,   // [B][T][U] queries (shifted decoder states)
        const bf16* __restrict__ H,     // [B][T][U] keys   ([N][K] layout for GEMM1 B)
        const bf16* __restrict__ Hu,    // [B][U][T] values ([N][K] layout for GEMM2 B)
        bf16* __restrict__ Aout){       // [B][T][U]
    __shared__ bf16 lds_sc[T_ * T_];    // 32 KB scores (bf16)
    __shared__ bf16 lds_p [T_ * T_];    // 32 KB probs  (bf16)
    int b = blockIdx.x;
    int tid = threadIdx.x, lane = tid & 31, w = tid >> 5;
    int ln = lane & 15, lh = lane >> 4;
    const bf16* Sb  = Ssh + (size_t)b * T_ * UNITS_;
    const bf16* Hb  = H   + (size_t)b * T_ * UNITS_;
    const bf16* Hub = Hu  + (size_t)b * UNITS_ * T_;

    {   // GEMM1: M=T(q), N=T(k), K=U; wave w -> N strip [16w,16w+16).
        // 8 independent accumulators -> dependent WMMAs 8 apart (no NOPs).
        int n0 = w * 16;
        const bf16* brow = Hb + (size_t)(n0 + ln) * UNITS_;
        v8f acc[8];
#pragma unroll
        for (int mt = 0; mt < 8; ++mt) acc[mt] = (v8f){0.f,0.f,0.f,0.f,0.f,0.f,0.f,0.f};
        for (int k0 = 0; k0 < UNITS_; k0 += 32){
            v16bf bf = load_b_frag(brow, k0, lane);
#pragma unroll
            for (int mt = 0; mt < 8; ++mt){
                v16bf a = load_a_frag(Sb + (size_t)(mt * 16 + ln) * UNITS_, k0, lane);
                acc[mt] = wmma_bf16(a, bf, acc[mt]);
            }
        }
#pragma unroll
        for (int mt = 0; mt < 8; ++mt)
#pragma unroll
            for (int r = 0; r < 8; ++r){
                int q = mt * 16 + (lh << 3) + r;
                lds_sc[q * T_ + n0 + ln] = (bf16)acc[mt][r];
            }
    }
    __syncthreads();

    if (tid < T_){   // softmax over k for row q=tid
        int q = tid;
        float mx = -1e30f;
        for (int k = 0; k < T_; ++k) mx = fmaxf(mx, (float)lds_sc[q * T_ + k]);
        float sum = 0.f;
        for (int k = 0; k < T_; ++k) sum += __expf((float)lds_sc[q * T_ + k] - mx);
        float inv = 1.0f / sum;
        for (int k = 0; k < T_; ++k)
            lds_p[q * T_ + k] = (bf16)(__expf((float)lds_sc[q * T_ + k] - mx) * inv);
    }
    __syncthreads();

    {   // GEMM2: M=T(q), N=U, K=T; wave w -> m-tile w; A-frags hoisted (K=128).
        // 4 n-tiles per iteration -> 4 independent accumulators, no hazard NOPs.
        int mt = w;
        v16bf afr[4];
#pragma unroll
        for (int kk = 0; kk < 4; ++kk)
            afr[kk] = load_a_frag(lds_p + (mt * 16 + ln) * T_, kk * 32, lane);
        for (int ng = 0; ng < UNITS_ / 64; ++ng){       // 8 groups of 4 n-tiles
            v8f acc[4];
            const bf16* brow[4];
#pragma unroll
            for (int j = 0; j < 4; ++j){
                acc[j]  = (v8f){0.f,0.f,0.f,0.f,0.f,0.f,0.f,0.f};
                brow[j] = Hub + (size_t)((ng * 4 + j) * 16 + ln) * T_;
            }
#pragma unroll
            for (int kk = 0; kk < 4; ++kk)
#pragma unroll
                for (int j = 0; j < 4; ++j)
                    acc[j] = wmma_bf16(afr[kk], load_b_frag(brow[j], kk * 32, lane), acc[j]);
#pragma unroll
            for (int j = 0; j < 4; ++j)
#pragma unroll
                for (int r = 0; r < 8; ++r){
                    int q = mt * 16 + (lh << 3) + r;
                    Aout[((size_t)b * T_ + q) * UNITS_ + (ng * 4 + j) * 16 + ln] = (bf16)acc[j][r];
                }
        }
    }
}

// ------------------------------------------- dense: logits = [S|A] @ W + b (fp32 -> d_out)
// grid (512 m-tiles, 8 n-chunks of 256), block 128. Each wave: 4 n-tiles with 4
// independent accumulators (no WMMA RAW NOPs) and 4x A-fragment reuse.
// Wt is padded to 2048 rows; stores guarded at the 125 real tiles.
__global__ void __launch_bounds__(128) k_dense(
        const bf16* __restrict__ S,     // [B*T][U]
        const bf16* __restrict__ A,     // [B*T][U]
        const bf16* __restrict__ Wt,    // [VTILE_PAD][2U] (rows >= VOCAB are pad)
        const float* __restrict__ bias, // [VOCAB]
        float* __restrict__ out){       // [B*T][VOCAB]
    int lane = threadIdx.x & 31, w = threadIdx.x >> 5;
    int ln = lane & 15, lh = lane >> 4;
    int mt  = blockIdx.x;
    int nt0 = blockIdx.y * 16 + w * 4;          // first of 4 n-tiles for this wave
    const bf16* srow = S + (size_t)(mt * 16 + ln) * UNITS_;
    const bf16* arow = A + (size_t)(mt * 16 + ln) * UNITS_;
    v8f acc[4];
    const bf16* brow[4];
#pragma unroll
    for (int j = 0; j < 4; ++j){
        acc[j]  = (v8f){0.f,0.f,0.f,0.f,0.f,0.f,0.f,0.f};
        brow[j] = Wt + (size_t)((nt0 + j) * 16 + ln) * (2 * UNITS_);
    }
    for (int k0 = 0; k0 < 2 * UNITS_; k0 += 32){
        v16bf af = (k0 < UNITS_) ? load_a_frag(srow, k0, lane)
                                 : load_a_frag(arow, k0 - UNITS_, lane);
#pragma unroll
        for (int j = 0; j < 4; ++j)
            acc[j] = wmma_bf16(af, load_b_frag(brow[j], k0, lane), acc[j]);
    }
#pragma unroll
    for (int j = 0; j < 4; ++j){
        int nt = nt0 + j;
        if (nt < VOCAB_ / 16){                   // wave-uniform guard
            float bv = bias[nt * 16 + ln];
#pragma unroll
            for (int r = 0; r < 8; ++r){
                int m = mt * 16 + (lh << 3) + r;
                out[(size_t)m * VOCAB_ + nt * 16 + ln] = acc[j][r] + bv;
            }
        }
    }
}

// in-place row softmax over VOCAB on d_out
__global__ void __launch_bounds__(256) k_softmax(float* __restrict__ out){
    __shared__ float red[256];
    int tid = threadIdx.x;
    float* p = out + (size_t)blockIdx.x * VOCAB_;
    float mx = -1e30f;
    for (int i = tid; i < VOCAB_; i += 256) mx = fmaxf(mx, p[i]);
    red[tid] = mx; __syncthreads();
    for (int s = 128; s > 0; s >>= 1){ if (tid < s) red[tid] = fmaxf(red[tid], red[tid + s]); __syncthreads(); }
    mx = red[0]; __syncthreads();
    float sum = 0.f;
    for (int i = tid; i < VOCAB_; i += 256) sum += __expf(p[i] - mx);
    red[tid] = sum; __syncthreads();
    for (int s = 128; s > 0; s >>= 1){ if (tid < s) red[tid] += red[tid + s]; __syncthreads(); }
    float inv = 1.0f / red[0]; __syncthreads();
    for (int i = tid; i < VOCAB_; i += 256) p[i] = __expf(p[i] - mx) * inv;
}

// ---------------------------------------------------------------------- launch
extern "C" void kernel_launch(void* const* d_in, const int* in_sizes, int n_in,
                              void* d_out, int out_size, void* d_ws, size_t ws_size,
                              hipStream_t stream){
    (void)in_sizes; (void)n_in; (void)out_size; (void)ws_size;
    const int*   x      = (const int*)  d_in[0];
    const int*   y      = (const int*)  d_in[1];
    const float* encEmb = (const float*)d_in[2];
    const float* encWx  = (const float*)d_in[3];
    const float* encWh  = (const float*)d_in[4];
    const float* encB   = (const float*)d_in[5];
    const float* decEmb = (const float*)d_in[6];
    const float* decWx  = (const float*)d_in[7];
    const float* decWh  = (const float*)d_in[8];
    const float* decB   = (const float*)d_in[9];
    const float* denW   = (const float*)d_in[10];
    const float* denB   = (const float*)d_in[11];
    float* out = (float*)d_out;

    char* ws = (char*)d_ws;
    size_t off = 0;
    auto carve = [&](size_t bytes) -> char* {
        char* p = ws + off;
        off = (off + bytes + 255) & ~(size_t)255;
        return p;
    };
    bf16* encEmbB = (bf16*)carve((size_t)VOCAB_ * EMB_ * 2);
    bf16* decEmbB = (bf16*)carve((size_t)VOCAB_ * EMB_ * 2);
    bf16* encWxt  = (bf16*)carve((size_t)GATES_ * EMB_ * 2);
    bf16* decWxt  = (bf16*)carve((size_t)GATES_ * EMB_ * 2);
    bf16* encWht  = (bf16*)carve((size_t)GATES_ * UNITS_ * 2);
    bf16* decWht  = (bf16*)carve((size_t)GATES_ * UNITS_ * 2);
    bf16* denWt   = (bf16*)carve((size_t)VTILE_PAD_ * 2 * UNITS_ * 2); // padded rows
    float* encXz  = (float*)carve((size_t)B_ * T_ * GATES_ * 4);
    float* decXz  = (float*)carve((size_t)B_ * T_ * GATES_ * 4);
    bf16* Hbf     = (bf16*)carve((size_t)B_ * T_ * UNITS_ * 2);
    bf16* Hu      = (bf16*)carve((size_t)B_ * UNITS_ * T_ * 2);
    bf16* Sbf     = (bf16*)carve((size_t)B_ * T_ * UNITS_ * 2);
    bf16* Sshift  = (bf16*)carve((size_t)B_ * T_ * UNITS_ * 2);
    bf16* Abf     = (bf16*)carve((size_t)B_ * T_ * UNITS_ * 2);
    float* hEnc   = (float*)carve((size_t)B_ * UNITS_ * 4);
    float* cEnc   = (float*)carve((size_t)B_ * UNITS_ * 4);
    float* hDump  = (float*)carve((size_t)B_ * UNITS_ * 4);
    float* cDec   = (float*)carve((size_t)B_ * UNITS_ * 4);

    auto cdiv = [](int a, int b){ return (a + b - 1) / b; };

    // 1) weight prep
    k_convert_bf16<<<cdiv(VOCAB_ * EMB_, 256), 256, 0, stream>>>(encEmb, encEmbB, VOCAB_ * EMB_);
    k_convert_bf16<<<cdiv(VOCAB_ * EMB_, 256), 256, 0, stream>>>(decEmb, decEmbB, VOCAB_ * EMB_);
    k_transpose_bf16<<<cdiv(EMB_ * GATES_, 256), 256, 0, stream>>>(encWx, encWxt, EMB_, GATES_);
    k_transpose_bf16<<<cdiv(EMB_ * GATES_, 256), 256, 0, stream>>>(decWx, decWxt, EMB_, GATES_);
    k_transpose_bf16<<<cdiv(UNITS_ * GATES_, 256), 256, 0, stream>>>(encWh, encWht, UNITS_, GATES_);
    k_transpose_bf16<<<cdiv(UNITS_ * GATES_, 256), 256, 0, stream>>>(decWh, decWht, UNITS_, GATES_);
    k_transpose_bf16<<<cdiv(2 * UNITS_ * VOCAB_, 256), 256, 0, stream>>>(denW, denWt, 2 * UNITS_, VOCAB_);

    // 2) input-projection GEMMs
    dim3 gXZ(B_ * T_ / 16, GATES_ / 64);
    k_xz<<<gXZ, 128, 0, stream>>>(x, encEmbB, encWxt, encB, encXz);
    k_xz<<<gXZ, 128, 0, stream>>>(y, decEmbB, decWxt, decB, decXz);

    // 3) encoder recurrence (single WGP-resident workgroup, 32 waves)
    k_lstm_scan<<<1, 1024, 0, stream>>>(encXz, encWht, nullptr, nullptr,
                                        cEnc, hEnc, Hbf, nullptr);
    // 4) H transpose for attention value GEMM
    k_transpose_H<<<cdiv(B_ * T_ * UNITS_, 256), 256, 0, stream>>>(Hbf, Hu);

    // 5) decoder recurrence (h0,c0 = encoder finals; also emits shifted queries)
    k_lstm_scan<<<1, 1024, 0, stream>>>(decXz, decWht, hEnc, cEnc,
                                        cDec, hDump, Sbf, Sshift);

    // 6) fused Luong attention per batch
    k_attention<<<B_, 256, 0, stream>>>(Sshift, Hbf, Hu, Abf);

    // 7) dense logits + softmax (in-place on d_out)
    dim3 gD(B_ * T_ / 16, 8);
    k_dense<<<gD, 128, 0, stream>>>(Sbf, Abf, denWt, denB, out);
    k_softmax<<<B_ * T_, 256, 0, stream>>>(out);
}
